// EncoderProcesserDecoder_60653528154680
// MI455X (gfx1250) — matrix operand order repaired and checked by reading, not compiled, not run on hardware
//
#include <hip/hip_runtime.h>

// ============================================================================
// MeshGraphNets-style Encoder -> 15x Processor -> Decoder GNN for MI455X.
//
// Roofline: E buffer (600000x128 f32 = 307MB) streams from HBM; X/AGG (51MB
// each) live mostly in the 192MB L2, so x[sender]/x[receiver] gathers are
// cheap. With the e_old LDS cache, E moves 1 read + 1 write per layer
// (~614MB/layer, ~9.2GB total) -> ~0.5ms @ 23.3TB/s. The ~2 TFLOP of MLP
// matmuls run on v_wmma_f32_16x16x32_bf16 (bf16 A/B, f32 accumulate;
// LayerNorm after every MLP contains bf16 operand error), keeping compute
// under the memory time. Residual / LN state stays fp32.
//
// Per block: 256 threads = 8 wave32, 32 rows. Wave w owns output cols
// [16w,16w+16) and holds two 16x16 f32 accumulator tiles (rows 0-15, 16-31).
// Activations ping-pong through LDS as bf16. Weights are repacked once per
// call into the per-lane WMMA B-fragment layout (one contiguous 32B load).
// ============================================================================

typedef __attribute__((ext_vector_type(16))) __bf16 v16bf;
typedef __attribute__((ext_vector_type(8)))  float  v8f;

#define HID     128
#define NLAYERS 15
#define NNODES  100000
#define NEDGES  600000

union PK4 { __bf16 h[4]; uint2 u; };

// ---------------------------------------------------------------------------
// Weight pack: row-major f32 W[K][N] -> bf16 B-fragment layout.
// ISA 16-bit B (32x16): lane<16 holds col=lane, K = kt*32 + {0..7}U{16..23};
// lane>=16 holds col=lane-16, K = kt*32 + {8..15}U{24..31}.
// dst index = ((((l*KB + kt)*NB + nt)*32 + lane)*16 + j)
// ---------------------------------------------------------------------------
__global__ __launch_bounds__(256) void pack_w(const float* __restrict__ src,
                                              __bf16* __restrict__ dst,
                                              int Ksrc, int Kpad, int N, int L)
{
    const int NB = N >> 4, KB = Kpad >> 5;
    const long total = (long)L * Kpad * N;
    for (long i = (long)blockIdx.x * blockDim.x + threadIdx.x; i < total;
         i += (long)gridDim.x * blockDim.x) {
        int  j    = (int)(i & 15);
        int  lane = (int)((i >> 4) & 31);
        long t    = i >> 9;                 // (l*KB + kt)*NB + nt
        int  nt   = (int)(t % NB);
        long t2   = t / NB;
        int  kt   = (int)(t2 % KB);
        int  l    = (int)(t2 / KB);
        int  n    = nt * 16 + (lane & 15);
        int  k    = kt * 32 + ((lane >> 4) << 3) + (j & 7) + ((j >> 3) << 4);
        float v   = (k < Ksrc) ? src[(size_t)l * Ksrc * N + (size_t)k * N + n] : 0.f;
        dst[i] = (__bf16)v;
    }
}

// ---------------------------------------------------------------------------
// Fragment loads.
// A (16-bit, 16x32): lane<16 -> row=lane,     K {0..7}U{16..23}
//                    lane>=16 -> row=lane-16, K {8..15}U{24..31}
// ---------------------------------------------------------------------------
__device__ __forceinline__ v16bf ldA(const __bf16* p)
{
    union { uint4 q[2]; v16bf v; } u;
    u.q[0] = *(const uint4*)(p);
    u.q[1] = *(const uint4*)(p + 16);
    return u.v;
}
__device__ __forceinline__ v16bf ldB(const __bf16* p)
{
    union { uint4 q[2]; v16bf v; } u;
    u.q[0] = *(const uint4*)(p);
    u.q[1] = *(const uint4*)(p + 8);    // 32 contiguous bytes
    return u.v;
}

// One GEMM stage: act [32][kpad] bf16 (LDS) x packed W -> two 16x16 f32 tiles.
__device__ __forceinline__ void gemm32(const __bf16* __restrict__ act, int kpad,
                                       int kblocks, const __bf16* __restrict__ Wp,
                                       int wv, int lane, v8f& acc0, v8f& acc1)
{
    acc0 = v8f{0.f,0.f,0.f,0.f,0.f,0.f,0.f,0.f};
    acc1 = acc0;
    const int arow = lane & 15;
    const int koff = (lane >> 4) << 3;
    const __bf16* a0 = act + arow * kpad + koff;
    const __bf16* a1 = act + (arow + 16) * kpad + koff;
    const __bf16* bp = Wp + (size_t)(wv * 32 + lane) * 16;
    for (int kt = 0; kt < kblocks; ++kt) {
        v16bf B  = ldB(bp);
        v16bf A0 = ldA(a0);
        v16bf A1 = ldA(a1);
        acc0 = __builtin_amdgcn_wmma_f32_16x16x32_bf16(false, A0, false, B,
                                                       (short)0, acc0, false, false);
        acc1 = __builtin_amdgcn_wmma_f32_16x16x32_bf16(false, A1, false, B,
                                                       (short)0, acc1, false, false);
        a0 += 32; a1 += 32; bp += 4096;   // next K-block of packed weights
    }
}

// C/D layout: VGPR r, lane<16 -> (M=r, N=lane); lane>=16 -> (M=r+8, N=lane-16)
__device__ __forceinline__ void epi_relu_bf16(const v8f& acc0, const v8f& acc1,
                                              const float* __restrict__ bias,
                                              __bf16* __restrict__ out, int wv, int lane)
{
    const int col = wv * 16 + (lane & 15);
    const float b = bias[col];
    const int rb = (lane >> 4) << 3;
#pragma unroll
    for (int r = 0; r < 8; ++r) {
        float v0 = fmaxf(acc0[r] + b, 0.f);
        float v1 = fmaxf(acc1[r] + b, 0.f);
        out[(rb + r) * HID + col]      = (__bf16)v0;
        out[(16 + rb + r) * HID + col] = (__bf16)v1;
    }
}
__device__ __forceinline__ void epi_f32(const v8f& acc0, const v8f& acc1,
                                        const float* __restrict__ bias,
                                        float* __restrict__ out, int wv, int lane)
{
    const int col = wv * 16 + (lane & 15);
    const float b = bias[col];
    const int rb = (lane >> 4) << 3;
#pragma unroll
    for (int r = 0; r < 8; ++r) {
        out[(rb + r) * HID + col]      = acc0[r] + b;
        out[(16 + rb + r) * HID + col] = acc1[r] + b;
    }
}

// 4-stage MLP: sA(kpad1) -> relu -> 128 -> relu -> 128 -> relu -> 128 -> sF(f32)
__device__ __forceinline__ void run_mlp4(const __bf16* sA, int kpad1,
    __bf16* sB, __bf16* sC, float* sF,
    const __bf16* W1p, const __bf16* W2p, const __bf16* W3p, const __bf16* W4p,
    const float* b1, const float* b2, const float* b3, const float* b4,
    int wv, int lane)
{
    v8f a0, a1;
    gemm32(sA, kpad1, kpad1 >> 5, W1p, wv, lane, a0, a1);
    epi_relu_bf16(a0, a1, b1, sB, wv, lane);  __syncthreads();
    gemm32(sB, HID, 4, W2p, wv, lane, a0, a1);
    epi_relu_bf16(a0, a1, b2, sC, wv, lane);  __syncthreads();
    gemm32(sC, HID, 4, W3p, wv, lane, a0, a1);
    epi_relu_bf16(a0, a1, b3, sB, wv, lane);  __syncthreads();
    gemm32(sB, HID, 4, W4p, wv, lane, a0, a1);
    epi_f32(a0, a1, b4, sF, wv, lane);        __syncthreads();
}

__device__ __forceinline__ void ln_stats(const float* sF, float* s_mu, float* s_rs, int tid)
{
    if (tid < 32) {
        const float* r = sF + tid * HID;
        float s = 0.f, s2 = 0.f;
        for (int c = 0; c < HID; ++c) { float v = r[c]; s += v; s2 += v * v; }
        float mu  = s  * (1.f / HID);
        float var = s2 * (1.f / HID) - mu * mu;
        s_mu[tid] = mu;
        s_rs[tid] = rsqrtf(var + 1e-5f);
    }
    __syncthreads();
}

// ---------------------------------------------------------------------------
// Encoder (node or edge): raw attrs (incols<=32, zero-padded) -> MLP+LN -> dst
// ---------------------------------------------------------------------------
__global__ __launch_bounds__(256) void enc_kernel(
    const float* __restrict__ src, int incols,
    const __bf16* __restrict__ W1p, const __bf16* __restrict__ W2p,
    const __bf16* __restrict__ W3p, const __bf16* __restrict__ W4p,
    const float* __restrict__ b1, const float* __restrict__ b2,
    const float* __restrict__ b3, const float* __restrict__ b4,
    const float* __restrict__ g,  const float* __restrict__ beta,
    float* __restrict__ dst)
{
    __shared__ __align__(16) __bf16 sA[32 * 32];
    __shared__ __align__(16) __bf16 sB[32 * HID];
    __shared__ __align__(16) __bf16 sC[32 * HID];
    __shared__ __align__(16) float  sF[32 * HID];
    __shared__ float s_mu[32], s_rs[32];
    const int tid = threadIdx.x, wv = tid >> 5, lane = tid & 31;
    const long base = (long)blockIdx.x * 32;

    for (int idx = tid; idx < 32 * 32; idx += 256) {
        int row = idx >> 5, c = idx & 31;
        float v = (c < incols) ? src[(base + row) * incols + c] : 0.f;
        sA[idx] = (__bf16)v;
    }
    __syncthreads();
    run_mlp4(sA, 32, sB, sC, sF, W1p, W2p, W3p, W4p, b1, b2, b3, b4, wv, lane);
    ln_stats(sF, s_mu, s_rs, tid);
    for (int q = tid; q < 32 * 32; q += 256) {
        int row = q >> 5, c4 = (q & 31) << 2;
        float mu = s_mu[row], rs = s_rs[row];
        float4 f  = *(const float4*)&sF[row * HID + c4];
        float4 gg = *(const float4*)&g[c4];
        float4 bb = *(const float4*)&beta[c4];
        float4 o  = make_float4((f.x - mu) * rs * gg.x + bb.x,
                                (f.y - mu) * rs * gg.y + bb.y,
                                (f.z - mu) * rs * gg.z + bb.z,
                                (f.w - mu) * rs * gg.w + bb.w);
        *(float4*)&dst[(base + row) * HID + c4] = o;
    }
}

// ---------------------------------------------------------------------------
// Edge block: concat(x[s], x[r], e)[384] -> MLP+LN = e_new;
// E = e_old + e_new (e_old cached in LDS -> single HBM read of E);
// atomic scatter e_new into AGG[receiver].
// ---------------------------------------------------------------------------
__global__ __launch_bounds__(256) void edge_proc_kernel(
    const float* __restrict__ X, float* __restrict__ E,
    const int* __restrict__ eidx, int nE, float* __restrict__ AGG,
    const __bf16* __restrict__ W1p, const __bf16* __restrict__ W2p,
    const __bf16* __restrict__ W3p, const __bf16* __restrict__ W4p,
    const float* __restrict__ b1, const float* __restrict__ b2,
    const float* __restrict__ b3, const float* __restrict__ b4,
    const float* __restrict__ g,  const float* __restrict__ beta)
{
    __shared__ __align__(16) __bf16 sA[32 * 384];
    __shared__ __align__(16) __bf16 sB[32 * HID];
    __shared__ __align__(16) __bf16 sC[32 * HID];
    __shared__ __align__(16) float  sF[32 * HID];
    __shared__ __align__(16) float  sEold[32 * HID];
    __shared__ float s_mu[32], s_rs[32];
    __shared__ int s_snd[32], s_rcv[32];
    const int tid = threadIdx.x, wv = tid >> 5, lane = tid & 31;
    const long base = (long)blockIdx.x * 32;

    if (tid < 32) {
        s_snd[tid] = eidx[base + tid];
        s_rcv[tid] = eidx[(long)nE + base + tid];
    }
    __syncthreads();
    for (int q = tid; q < 32 * 96; q += 256) {        // 96 float4 per row
        int row = q / 96, c4 = (q - row * 96) << 2;
        float4 v;
        if (c4 < 128)      v = *(const float4*)&X[(long)s_snd[row] * HID + c4];
        else if (c4 < 256) v = *(const float4*)&X[(long)s_rcv[row] * HID + (c4 - 128)];
        else {
            v = *(const float4*)&E[(base + row) * HID + (c4 - 256)];
            *(float4*)&sEold[row * HID + (c4 - 256)] = v;
        }
        PK4 pk;
        pk.h[0] = (__bf16)v.x; pk.h[1] = (__bf16)v.y;
        pk.h[2] = (__bf16)v.z; pk.h[3] = (__bf16)v.w;
        *(uint2*)&sA[row * 384 + c4] = pk.u;
    }
    __syncthreads();
    run_mlp4(sA, 384, sB, sC, sF, W1p, W2p, W3p, W4p, b1, b2, b3, b4, wv, lane);
    ln_stats(sF, s_mu, s_rs, tid);
    for (int q = tid; q < 32 * 32; q += 256) {
        int row = q >> 5, c4 = (q & 31) << 2;
        float mu = s_mu[row], rs = s_rs[row];
        float4 f  = *(const float4*)&sF[row * HID + c4];
        float4 gg = *(const float4*)&g[c4];
        float4 bb = *(const float4*)&beta[c4];
        float v0 = (f.x - mu) * rs * gg.x + bb.x;
        float v1 = (f.y - mu) * rs * gg.y + bb.y;
        float v2 = (f.z - mu) * rs * gg.z + bb.z;
        float v3 = (f.w - mu) * rs * gg.w + bb.w;
        float4 eo = *(const float4*)&sEold[row * HID + c4];
        long ei = base + row;
        *(float4*)&E[ei * HID + c4] =
            make_float4(eo.x + v0, eo.y + v1, eo.z + v2, eo.w + v3);
        float* ap = &AGG[(long)s_rcv[row] * HID + c4];
        unsafeAtomicAdd(ap + 0, v0);
        unsafeAtomicAdd(ap + 1, v1);
        unsafeAtomicAdd(ap + 2, v2);
        unsafeAtomicAdd(ap + 3, v3);
    }
}

// ---------------------------------------------------------------------------
// Node block: concat(x, agg)[256] -> MLP+LN = x_new; X = x_old + x_new.
// ---------------------------------------------------------------------------
__global__ __launch_bounds__(256) void node_proc_kernel(
    float* __restrict__ X, const float* __restrict__ AGG,
    const __bf16* __restrict__ W1p, const __bf16* __restrict__ W2p,
    const __bf16* __restrict__ W3p, const __bf16* __restrict__ W4p,
    const float* __restrict__ b1, const float* __restrict__ b2,
    const float* __restrict__ b3, const float* __restrict__ b4,
    const float* __restrict__ g,  const float* __restrict__ beta)
{
    __shared__ __align__(16) __bf16 sA[32 * 256];
    __shared__ __align__(16) __bf16 sB[32 * HID];
    __shared__ __align__(16) __bf16 sC[32 * HID];
    __shared__ __align__(16) float  sF[32 * HID];
    __shared__ __align__(16) float  sXold[32 * HID];
    __shared__ float s_mu[32], s_rs[32];
    const int tid = threadIdx.x, wv = tid >> 5, lane = tid & 31;
    const long base = (long)blockIdx.x * 32;

    for (int q = tid; q < 32 * 64; q += 256) {        // 64 float4 per row
        int row = q >> 6, c4 = (q & 63) << 2;
        float4 v;
        if (c4 < 128) {
            v = *(const float4*)&X[(base + row) * HID + c4];
            *(float4*)&sXold[row * HID + c4] = v;
        } else {
            v = *(const float4*)&AGG[(base + row) * HID + (c4 - 128)];
        }
        PK4 pk;
        pk.h[0] = (__bf16)v.x; pk.h[1] = (__bf16)v.y;
        pk.h[2] = (__bf16)v.z; pk.h[3] = (__bf16)v.w;
        *(uint2*)&sA[row * 256 + c4] = pk.u;
    }
    __syncthreads();
    run_mlp4(sA, 256, sB, sC, sF, W1p, W2p, W3p, W4p, b1, b2, b3, b4, wv, lane);
    ln_stats(sF, s_mu, s_rs, tid);
    for (int q = tid; q < 32 * 32; q += 256) {
        int row = q >> 5, c4 = (q & 31) << 2;
        float mu = s_mu[row], rs = s_rs[row];
        float4 f  = *(const float4*)&sF[row * HID + c4];
        float4 gg = *(const float4*)&g[c4];
        float4 bb = *(const float4*)&beta[c4];
        float4 xo = *(const float4*)&sXold[row * HID + c4];
        *(float4*)&X[(base + row) * HID + c4] =
            make_float4(xo.x + (f.x - mu) * rs * gg.x + bb.x,
                        xo.y + (f.y - mu) * rs * gg.y + bb.y,
                        xo.z + (f.z - mu) * rs * gg.z + bb.z,
                        xo.w + (f.w - mu) * rs * gg.w + bb.w);
    }
}

// ---------------------------------------------------------------------------
// Decoder: 3 WMMA stages (relu) then 128->3 dot products (f32 weights).
// ---------------------------------------------------------------------------
__global__ __launch_bounds__(256) void dec_kernel(
    const float* __restrict__ X,
    const __bf16* __restrict__ W1p, const __bf16* __restrict__ W2p,
    const __bf16* __restrict__ W3p,
    const float* __restrict__ b1, const float* __restrict__ b2,
    const float* __restrict__ b3,
    const float* __restrict__ W4, const float* __restrict__ b4,
    float* __restrict__ out)
{
    __shared__ __align__(16) __bf16 sA[32 * HID];
    __shared__ __align__(16) __bf16 sB[32 * HID];
    __shared__ __align__(16) __bf16 sC[32 * HID];
    const int tid = threadIdx.x, wv = tid >> 5, lane = tid & 31;
    const long base = (long)blockIdx.x * 32;

    for (int q = tid; q < 32 * 32; q += 256) {
        int row = q >> 5, c4 = (q & 31) << 2;
        float4 v = *(const float4*)&X[(base + row) * HID + c4];
        PK4 pk;
        pk.h[0] = (__bf16)v.x; pk.h[1] = (__bf16)v.y;
        pk.h[2] = (__bf16)v.z; pk.h[3] = (__bf16)v.w;
        *(uint2*)&sA[row * HID + c4] = pk.u;
    }
    __syncthreads();
    v8f a0, a1;
    gemm32(sA, HID, 4, W1p, wv, lane, a0, a1);
    epi_relu_bf16(a0, a1, b1, sB, wv, lane);  __syncthreads();
    gemm32(sB, HID, 4, W2p, wv, lane, a0, a1);
    epi_relu_bf16(a0, a1, b2, sC, wv, lane);  __syncthreads();
    gemm32(sC, HID, 4, W3p, wv, lane, a0, a1);
    epi_relu_bf16(a0, a1, b3, sB, wv, lane);  __syncthreads();
    if (tid < 96) {
        int row = tid / 3, o = tid - row * 3;
        float s = b4[o];
        for (int c = 0; c < HID; ++c)
            s += (float)sB[row * HID + c] * W4[c * 3 + o];
        out[(base + row) * 3 + o] = s;
    }
}

// ===========================================================================
// Host side. The nested-param flattening convention is auto-detected from
// in_sizes (works for both recursive insertion-order and sorted-key order):
//   * member order: in_sizes[block+1]==128 -> (W1,b1,W2,b2,...,g,beta)
//                   else                  -> (W1..W4,b1..b4,beta,g)
//   * enc 'node' vs 'edge' first: W1 leaf 11*128 vs 3*128
//   * proc 'edge' vs 'node' first: W1 leaf 15*384*128 vs 15*256*128
// ===========================================================================
extern "C" void kernel_launch(void* const* d_in, const int* in_sizes, int n_in,
                              void* d_out, int out_size, void* d_ws, size_t ws_size,
                              hipStream_t stream)
{
    (void)n_in; (void)out_size; (void)ws_size;
    const float* node_attr = (const float*)d_in[0];
    const float* edge_attr = (const float*)d_in[1];
    const int*   eidx      = (const int*)d_in[2];
    auto F = [&](int i) { return (const float*)d_in[i]; };

    // ---- input-layout auto-detection
    int mW1, mW2, mW3, mW4, mb1, mb2, mb3, mb4, mg, mbeta;
    if (in_sizes[4] == 128) {           // insertion order: W1,b1,W2,b2,...
        mW1 = 0; mb1 = 1; mW2 = 2; mb2 = 3; mW3 = 4; mb3 = 5;
        mW4 = 6; mb4 = 7; mg = 8; mbeta = 9;
    } else {                            // sorted keys: W1..W4,b1..b4,beta,g
        mW1 = 0; mW2 = 1; mW3 = 2; mW4 = 3; mb1 = 4; mb2 = 5;
        mb3 = 6; mb4 = 7; mbeta = 8; mg = 9;
    }
    const int ENC_N  = (in_sizes[3] == 11 * 128) ? 3 : 13;
    const int ENC_E  = (ENC_N == 3) ? 13 : 3;
    const int PROC_E = (in_sizes[23] == NLAYERS * 384 * 128) ? 23 : 33;
    const int PROC_N = (PROC_E == 23) ? 33 : 23;
    const int DEC    = 43;

    // ---- workspace carve (X 51.2MB, E 307.2MB, AGG 51.2MB, packed W 5.7MB)
    char* p = (char*)d_ws;
    auto carve = [&](size_t bytes) { void* r = (void*)p;
                                     p += (bytes + 255) & ~(size_t)255; return r; };
    float*  X    = (float*)carve((size_t)NNODES * HID * 4);
    float*  E    = (float*)carve((size_t)NEDGES * HID * 4);
    float*  AGG  = (float*)carve((size_t)NNODES * HID * 4);
    __bf16* pool = (__bf16*)carve((size_t)2859008 * 2);

    size_t o = 0;
    auto take = [&](size_t n) { size_t r = o; o += n; return r; };
    const size_t W128 = 128 * 128;
    const size_t o_enW1 = take(32 * 128),  o_enW2 = take(W128),
                 o_enW3 = take(W128),      o_enW4 = take(W128);
    const size_t o_eeW1 = take(32 * 128),  o_eeW2 = take(W128),
                 o_eeW3 = take(W128),      o_eeW4 = take(W128);
    const size_t o_peW1 = take((size_t)NLAYERS * 384 * 128),
                 o_peW2 = take((size_t)NLAYERS * W128),
                 o_peW3 = take((size_t)NLAYERS * W128),
                 o_peW4 = take((size_t)NLAYERS * W128);
    const size_t o_pnW1 = take((size_t)NLAYERS * 256 * 128),
                 o_pnW2 = take((size_t)NLAYERS * W128),
                 o_pnW3 = take((size_t)NLAYERS * W128),
                 o_pnW4 = take((size_t)NLAYERS * W128);
    const size_t o_dW1  = take(W128), o_dW2 = take(W128), o_dW3 = take(W128);

    auto pack = [&](const float* src, size_t dstoff, int Ksrc, int Kpad, int L) {
        long total = (long)L * Kpad * 128;
        int blocks = (int)((total + 255) / 256);
        if (blocks > 4096) blocks = 4096;
        pack_w<<<blocks, 256, 0, stream>>>(src, pool + dstoff, Ksrc, Kpad, 128, L);
    };
    pack(F(ENC_N + mW1), o_enW1, 11, 32, 1);
    pack(F(ENC_N + mW2), o_enW2, 128, 128, 1);
    pack(F(ENC_N + mW3), o_enW3, 128, 128, 1);
    pack(F(ENC_N + mW4), o_enW4, 128, 128, 1);
    pack(F(ENC_E + mW1), o_eeW1, 3, 32, 1);
    pack(F(ENC_E + mW2), o_eeW2, 128, 128, 1);
    pack(F(ENC_E + mW3), o_eeW3, 128, 128, 1);
    pack(F(ENC_E + mW4), o_eeW4, 128, 128, 1);
    pack(F(PROC_E + mW1), o_peW1, 384, 384, NLAYERS);
    pack(F(PROC_E + mW2), o_peW2, 128, 128, NLAYERS);
    pack(F(PROC_E + mW3), o_peW3, 128, 128, NLAYERS);
    pack(F(PROC_E + mW4), o_peW4, 128, 128, NLAYERS);
    pack(F(PROC_N + mW1), o_pnW1, 256, 256, NLAYERS);
    pack(F(PROC_N + mW2), o_pnW2, 128, 128, NLAYERS);
    pack(F(PROC_N + mW3), o_pnW3, 128, 128, NLAYERS);
    pack(F(PROC_N + mW4), o_pnW4, 128, 128, NLAYERS);
    pack(F(DEC + mW1), o_dW1, 128, 128, 1);
    pack(F(DEC + mW2), o_dW2, 128, 128, 1);
    pack(F(DEC + mW3), o_dW3, 128, 128, 1);

    // ---- encoders
    enc_kernel<<<NNODES / 32, 256, 0, stream>>>(node_attr, 11,
        pool + o_enW1, pool + o_enW2, pool + o_enW3, pool + o_enW4,
        F(ENC_N + mb1), F(ENC_N + mb2), F(ENC_N + mb3), F(ENC_N + mb4),
        F(ENC_N + mg), F(ENC_N + mbeta), X);
    enc_kernel<<<NEDGES / 32, 256, 0, stream>>>(edge_attr, 3,
        pool + o_eeW1, pool + o_eeW2, pool + o_eeW3, pool + o_eeW4,
        F(ENC_E + mb1), F(ENC_E + mb2), F(ENC_E + mb3), F(ENC_E + mb4),
        F(ENC_E + mg), F(ENC_E + mbeta), E);

    // ---- 15 message-passing layers
    for (int l = 0; l < NLAYERS; ++l) {
        hipMemsetAsync(AGG, 0, (size_t)NNODES * HID * 4, stream);
        edge_proc_kernel<<<NEDGES / 32, 256, 0, stream>>>(X, E, eidx, NEDGES, AGG,
            pool + o_peW1 + (size_t)l * 384 * 128,
            pool + o_peW2 + (size_t)l * W128,
            pool + o_peW3 + (size_t)l * W128,
            pool + o_peW4 + (size_t)l * W128,
            F(PROC_E + mb1) + (size_t)l * HID, F(PROC_E + mb2) + (size_t)l * HID,
            F(PROC_E + mb3) + (size_t)l * HID, F(PROC_E + mb4) + (size_t)l * HID,
            F(PROC_E + mg)  + (size_t)l * HID, F(PROC_E + mbeta) + (size_t)l * HID);
        node_proc_kernel<<<NNODES / 32, 256, 0, stream>>>(X, AGG,
            pool + o_pnW1 + (size_t)l * 256 * 128,
            pool + o_pnW2 + (size_t)l * W128,
            pool + o_pnW3 + (size_t)l * W128,
            pool + o_pnW4 + (size_t)l * W128,
            F(PROC_N + mb1) + (size_t)l * HID, F(PROC_N + mb2) + (size_t)l * HID,
            F(PROC_N + mb3) + (size_t)l * HID, F(PROC_N + mb4) + (size_t)l * HID,
            F(PROC_N + mg)  + (size_t)l * HID, F(PROC_N + mbeta) + (size_t)l * HID);
    }

    // ---- decoder
    dec_kernel<<<NNODES / 32, 256, 0, stream>>>(X,
        pool + o_dW1, pool + o_dW2, pool + o_dW3,
        F(DEC + mb1), F(DEC + mb2), F(DEC + mb3),
        F(DEC + mW4), F(DEC + mb4), (float*)d_out);
}
